// LstmCellRudder_32968168964742
// MI455X (gfx1250) — matrix operand order repaired and checked by reading, not compiled, not run on hardware
//
#include <hip/hip_runtime.h>

// Problem constants
#define BDIM 256   // batch
#define TDIM 512   // time steps
#define NPOS 128
#define NACT 32
#define DDIM 160   // NPOS + NACT
#define HDIM 512   // hidden
#define G4H  2048  // 4*H
#define KBX  5     // k-blocks (of 32) covering x portion  (160/32)
#define KBH  16    // k-blocks covering h portion          (512/32)
#define KBT  21    // total k-blocks (672/32)

typedef __attribute__((ext_vector_type(16))) __bf16   v16bf;
typedef __attribute__((ext_vector_type(8)))  float    v8f;
typedef __attribute__((ext_vector_type(4)))  unsigned u32x4;
typedef __attribute__((ext_vector_type(8)))  unsigned u32x8;

// Build a 32-byte bf16 fragment from two 16-byte chunks.
static __device__ __forceinline__ v16bf ld_frag(const __bf16* p0, const __bf16* p1) {
    u32x4 lo = *(const u32x4*)p0;
    u32x4 hi = *(const u32x4*)p1;
    u32x8 t;
    t.lo = lo;
    t.hi = hi;
    return __builtin_bit_cast(v16bf, t);
}

static __device__ __forceinline__ v8f wmma_bf16(v16bf a, v16bf b, v8f c) {
    // (neg_a, A, neg_b, B, c_mod, C, reuse_a, reuse_b)
    return __builtin_amdgcn_wmma_f32_16x16x32_bf16(false, a, false, b, (short)0, c, false, false);
}

static __device__ __forceinline__ float sigmoidf_(float x) {
    return 1.0f / (1.0f + __expf(-x));
}
static __device__ __forceinline__ float tanhf_(float x) {
    float e = __expf(-2.0f * x);          // tanh(x) = (1-e)/(1+e), e = exp(-2x)
    return (1.0f - e) / (1.0f + e);
}

// ---------------------------------------------------------------------------
// Pack x = concat(obs, act) -> bf16, time-major [T, B, D]
// ---------------------------------------------------------------------------
__global__ void pack_x_kernel(const float* __restrict__ obs, const float* __restrict__ act,
                              __bf16* __restrict__ xPack) {
    int idx = blockIdx.x * blockDim.x + threadIdx.x;
    if (idx >= TDIM * BDIM * DDIM) return;
    int d = idx % DDIM;
    int b = (idx / DDIM) % BDIM;
    int t = idx / (DDIM * BDIM);
    float v = (d < NPOS) ? obs[((size_t)b * TDIM + t) * NPOS + d]
                         : act[((size_t)b * TDIM + t) * NACT + (d - NPOS)];
    xPack[idx] = (__bf16)v;
}

// ---------------------------------------------------------------------------
// Pack combined weight Wc[n, k] = k<160 ? W_ih[n,k] : W_hh[n,k-160], n in [0,2048)
// into B-fragment stream: elem index = ((nt*21 + kb)*32 + lane)*16 + j
//   lane 0-15 :  N = nt*16 + lane,     K = kb*32 + j (j<8), kb*32 + 8 + j (j>=8)
//   lane 16-31:  N = nt*16 + lane-16,  K = kb*32 + 8 + j (j<8), kb*32 + 16 + j (j>=8)
// ---------------------------------------------------------------------------
__global__ void pack_w_kernel(const float* __restrict__ Wih, const float* __restrict__ Whh,
                              __bf16* __restrict__ Wpack) {
    int idx = blockIdx.x * blockDim.x + threadIdx.x;
    if (idx >= G4H * (DDIM + HDIM)) return;
    int j    = idx & 15;
    int lane = (idx >> 4) & 31;
    int kb   = (idx >> 9) % KBT;
    int nt   = (idx >> 9) / KBT;
    int n = nt * 16 + (lane & 15);
    int k = kb * 32 + ((lane >> 4) << 3) + (j & 7) + ((j >> 3) << 4);
    float v = (k < DDIM) ? Wih[(size_t)n * DDIM + k]
                         : Whh[(size_t)n * HDIM + (k - DDIM)];
    Wpack[idx] = (__bf16)v;
}

// ---------------------------------------------------------------------------
// Init: h0 -> hHist slot 0 (bf16), c0 -> cBuf (f32), bias = b_ih + b_hh
// ---------------------------------------------------------------------------
__global__ void init_kernel(const float* __restrict__ h0, const float* __restrict__ c0,
                            const float* __restrict__ bih, const float* __restrict__ bhh,
                            __bf16* __restrict__ hHist, float* __restrict__ cBuf,
                            float* __restrict__ bias) {
    int i = blockIdx.x * blockDim.x + threadIdx.x;
    if (i < BDIM * HDIM) {
        hHist[i] = (__bf16)h0[i];
        cBuf[i]  = c0[i];
    }
    if (i < G4H) bias[i] = bih[i] + bhh[i];
}

// ---------------------------------------------------------------------------
// One LSTM timestep. 64 blocks x 128 threads (4 waves). Each wave owns a
// 32(batch) x 16(hidden) tile for all 4 gates: 8 WMMA accumulators, so each
// k-block issues 8 independent v_wmma against 12 b128 loads (1.5 loads/WMMA).
// ---------------------------------------------------------------------------
__global__ void __launch_bounds__(128) lstm_step_kernel(
    const __bf16* __restrict__ xPack, __bf16* __restrict__ hHist,
    const __bf16* __restrict__ Wpack, const float* __restrict__ bias,
    float* __restrict__ cBuf, int t)
{
    const int lane   = threadIdx.x & 31;
    const int wid    = threadIdx.x >> 5;
    const int tileId = blockIdx.x * 4 + wid;   // 0..255
    const int tile_m = tileId >> 5;            // 0..7   (batch/32)
    const int tile_n = tileId & 31;            // 0..31  (hidden/16)
    const int m0     = tile_m * 32;
    const int laneLo = lane & 15;
    const int laneHi = lane >> 4;

    // A-fragment rows (lanes 0-15: K+{0..7,16..23}; lanes 16-31: K+{8..15,24..31})
    const __bf16* xA0 = xPack + (size_t)t * BDIM * DDIM
                              + (size_t)(m0 + laneLo) * DDIM + laneHi * 8;
    const __bf16* xA1 = xA0 + (size_t)16 * DDIM;
    const __bf16* hA0 = hHist + (size_t)t * BDIM * HDIM
                              + (size_t)(m0 + laneLo) * HDIM + laneHi * 8;
    const __bf16* hA1 = hA0 + (size_t)16 * HDIM;

    // Per-gate packed-B streams (gate g covers hidden n-tile g*32 + tile_n)
    const __bf16* pB0 = Wpack + ((size_t)((0 * 32 + tile_n) * KBT) * 32 + lane) * 16;
    const __bf16* pB1 = Wpack + ((size_t)((1 * 32 + tile_n) * KBT) * 32 + lane) * 16;
    const __bf16* pB2 = Wpack + ((size_t)((2 * 32 + tile_n) * KBT) * 32 + lane) * 16;
    const __bf16* pB3 = Wpack + ((size_t)((3 * 32 + tile_n) * KBT) * 32 + lane) * 16;

    v8f acc[4][2];
    #pragma unroll
    for (int g = 0; g < 4; ++g)
        #pragma unroll
        for (int mt = 0; mt < 2; ++mt)
            acc[g][mt] = (v8f){};

    // K-loop part 1: x contribution (5 k-blocks of 32)
    #pragma unroll
    for (int kb = 0; kb < KBX; ++kb) {
        v16bf a0 = ld_frag(xA0 + kb * 32, xA0 + kb * 32 + 16);
        v16bf a1 = ld_frag(xA1 + kb * 32, xA1 + kb * 32 + 16);
        v16bf b0 = ld_frag(pB0, pB0 + 8); pB0 += 512;
        v16bf b1 = ld_frag(pB1, pB1 + 8); pB1 += 512;
        v16bf b2 = ld_frag(pB2, pB2 + 8); pB2 += 512;
        v16bf b3 = ld_frag(pB3, pB3 + 8); pB3 += 512;
        acc[0][0] = wmma_bf16(a0, b0, acc[0][0]);
        acc[0][1] = wmma_bf16(a1, b0, acc[0][1]);
        acc[1][0] = wmma_bf16(a0, b1, acc[1][0]);
        acc[1][1] = wmma_bf16(a1, b1, acc[1][1]);
        acc[2][0] = wmma_bf16(a0, b2, acc[2][0]);
        acc[2][1] = wmma_bf16(a1, b2, acc[2][1]);
        acc[3][0] = wmma_bf16(a0, b3, acc[3][0]);
        acc[3][1] = wmma_bf16(a1, b3, acc[3][1]);
    }
    // K-loop part 2: recurrent h contribution (16 k-blocks of 32)
    #pragma unroll
    for (int kb = 0; kb < KBH; ++kb) {
        v16bf a0 = ld_frag(hA0 + kb * 32, hA0 + kb * 32 + 16);
        v16bf a1 = ld_frag(hA1 + kb * 32, hA1 + kb * 32 + 16);
        v16bf b0 = ld_frag(pB0, pB0 + 8); pB0 += 512;
        v16bf b1 = ld_frag(pB1, pB1 + 8); pB1 += 512;
        v16bf b2 = ld_frag(pB2, pB2 + 8); pB2 += 512;
        v16bf b3 = ld_frag(pB3, pB3 + 8); pB3 += 512;
        acc[0][0] = wmma_bf16(a0, b0, acc[0][0]);
        acc[0][1] = wmma_bf16(a1, b0, acc[0][1]);
        acc[1][0] = wmma_bf16(a0, b1, acc[1][0]);
        acc[1][1] = wmma_bf16(a1, b1, acc[1][1]);
        acc[2][0] = wmma_bf16(a0, b2, acc[2][0]);
        acc[2][1] = wmma_bf16(a1, b2, acc[2][1]);
        acc[3][0] = wmma_bf16(a0, b3, acc[3][0]);
        acc[3][1] = wmma_bf16(a1, b3, acc[3][1]);
    }

    // Epilogue: C/D layout -> VGPR r, lane: M = base + r + 8*laneHi, N = tile_n*16 + laneLo
    const int n = tile_n * 16 + laneLo;
    const float bi = bias[n];
    const float bf = bias[HDIM + n];
    const float bg = bias[2 * HDIM + n];
    const float bo = bias[3 * HDIM + n];
    __bf16* hNext = hHist + (size_t)(t + 1) * BDIM * HDIM;

    #pragma unroll
    for (int mt = 0; mt < 2; ++mt) {
        #pragma unroll
        for (int r = 0; r < 8; ++r) {
            int m = m0 + mt * 16 + r + 8 * laneHi;
            size_t off = (size_t)m * HDIM + n;
            float iv = sigmoidf_(acc[0][mt][r] + bi);
            float fv = sigmoidf_(acc[1][mt][r] + bf);
            float gv = tanhf_(acc[2][mt][r] + bg);
            float ov = sigmoidf_(acc[3][mt][r] + bo);
            float cn = fv * cBuf[off] + iv * gv;
            cBuf[off]  = cn;
            hNext[off] = (__bf16)(ov * tanhf_(cn));
        }
    }
}

// ---------------------------------------------------------------------------
// y[b,t] = h_{t+1}[b,:] . W_out + b_out  — one wave per (b,t), deterministic.
// ---------------------------------------------------------------------------
__global__ void out_kernel(const __bf16* __restrict__ hHist, const float* __restrict__ Wout,
                           const float* __restrict__ bout, float* __restrict__ y) {
    int gw   = (blockIdx.x * blockDim.x + threadIdx.x) >> 5;
    int lane = threadIdx.x & 31;
    if (gw >= BDIM * TDIM) return;
    int b = gw >> 9;           // / TDIM
    int t = gw & (TDIM - 1);
    const __bf16* h = hHist + (size_t)(t + 1) * BDIM * HDIM + (size_t)b * HDIM;
    float s = 0.f;
    #pragma unroll 4
    for (int j = lane; j < HDIM; j += 32) s += (float)h[j] * Wout[j];
    #pragma unroll
    for (int o = 16; o > 0; o >>= 1) s += __shfl_xor(s, o, 32);
    if (lane == 0) y[(size_t)b * TDIM + t] = s + bout[0];
}

// ---------------------------------------------------------------------------
extern "C" void kernel_launch(void* const* d_in, const int* in_sizes, int n_in,
                              void* d_out, int out_size, void* d_ws, size_t ws_size,
                              hipStream_t stream) {
    const float* obs  = (const float*)d_in[0];
    const float* act  = (const float*)d_in[1];
    const float* Wih  = (const float*)d_in[2];
    const float* Whh  = (const float*)d_in[3];
    const float* bih  = (const float*)d_in[4];
    const float* bhh  = (const float*)d_in[5];
    const float* Wout = (const float*)d_in[6];
    const float* bout = (const float*)d_in[7];
    const float* h0   = (const float*)d_in[8];
    const float* c0   = (const float*)d_in[9];
    float* y = (float*)d_out;

    char* ws = (char*)d_ws;
    size_t off = 0;
    __bf16* hHist = (__bf16*)(ws + off); off += (size_t)(TDIM + 1) * BDIM * HDIM * 2; // 134.5 MB
    __bf16* xPack = (__bf16*)(ws + off); off += (size_t)TDIM * BDIM * DDIM * 2;       // 42 MB
    __bf16* Wpack = (__bf16*)(ws + off); off += (size_t)G4H * (DDIM + HDIM) * 2;      // 2.75 MB
    float*  cBuf  = (float*)(ws + off);  off += (size_t)BDIM * HDIM * 4;              // 0.5 MB
    float*  bias  = (float*)(ws + off);  off += (size_t)G4H * 4;

    pack_x_kernel<<<(TDIM * BDIM * DDIM + 255) / 256, 256, 0, stream>>>(obs, act, xPack);
    pack_w_kernel<<<(G4H * (DDIM + HDIM) + 255) / 256, 256, 0, stream>>>(Wih, Whh, Wpack);
    init_kernel<<<(BDIM * HDIM + 255) / 256, 256, 0, stream>>>(h0, c0, bih, bhh, hHist, cBuf, bias);

    for (int t = 0; t < TDIM; ++t) {
        lstm_step_kernel<<<64, 128, 0, stream>>>(xPack, hHist, Wpack, bias, cBuf, t);
    }

    out_kernel<<<(BDIM * TDIM * 32 + 255) / 256, 256, 0, stream>>>(hHist, Wout, bout, y);
}
